// TreeLSTMCell_83932250898829
// MI455X (gfx1250) — compile-verified, matrix operand order checked by reading, not run on hardware
//
#include <hip/hip_runtime.h>

#define DD 2048   // input size
#define HH 2048   // hidden size
#define KK 4096   // children

typedef __attribute__((ext_vector_type(16))) __bf16 v16bf;
typedef __attribute__((ext_vector_type(8)))  __bf16 v8bf;
typedef __attribute__((ext_vector_type(2)))  __bf16 v2bf;
typedef __attribute__((ext_vector_type(8)))  float  v8f;

// pack 2 fp32 -> packed 2xbf16. Prefer v_cvt_pk_bf16_f32 if declared; otherwise
// round-half-up (bits + 0x8000) + v_perm_b32 byte-pack: 3 VALU per 2 elements.
__device__ __forceinline__ unsigned pk2bf(float x, float y) {
#if __has_builtin(__builtin_amdgcn_cvt_pk_bf16_f32)
  union { v2bf b; unsigned u; } r;
  r.b = __builtin_amdgcn_cvt_pk_bf16_f32(x, y);
  return r.u;
#else
  unsigned rx = __float_as_uint(x) + 0x8000u;
  unsigned ry = __float_as_uint(y) + 0x8000u;
  // out = { ry[31:24], ry[23:16], rx[31:24], rx[23:16] }
  return __builtin_amdgcn_perm(ry, rx, 0x07060302u);
#endif
}

__device__ __forceinline__ uint2 pk4bf(float4 v) {
  uint2 r;
  r.x = pk2bf(v.x, v.y);
  r.y = pk2bf(v.z, v.w);
  return r;
}

__device__ __forceinline__ float sigmoidf_(float x) {
  return 1.0f / (1.0f + __expf(-x));
}

__device__ __forceinline__ float wave_reduce_add(float v) {
#pragma unroll
  for (int off = 16; off > 0; off >>= 1) v += __shfl_down(v, off, 32);
  return v;
}

// ---------------- h_sum = sum_k children_h[k,:] ----------------
__global__ __launch_bounds__(256) void hsum_part_kernel(
    const float* __restrict__ ch_h, float* __restrict__ part) {
  int j  = blockIdx.x * 256 + threadIdx.x;  // column
  int kc = blockIdx.y;                      // chunk of 512 rows
  float acc = 0.f;
  const float* p = ch_h + (size_t)kc * 512 * HH + j;
  for (int k = 0; k < 512; ++k) acc += p[(size_t)k * HH];
  part[(size_t)kc * HH + j] = acc;
}

__global__ __launch_bounds__(256) void hsum_reduce_kernel(
    const float* __restrict__ part, float* __restrict__ hsum) {
  int j = blockIdx.x * 256 + threadIdx.x;
  float s = 0.f;
#pragma unroll
  for (int kc = 0; kc < 8; ++kc) s += part[(size_t)kc * HH + j];
  hsum[j] = s;
}

// ---------------- fxb = W_fx @ x + b_fx + b_fh  ----------------
__global__ __launch_bounds__(256) void gemv_fx_kernel(
    const float* __restrict__ W_fx, const float* __restrict__ b_fx,
    const float* __restrict__ b_fh, const float* __restrict__ x,
    float* __restrict__ fxb) {
  int wave = threadIdx.x >> 5, lane = threadIdx.x & 31;
  int row = blockIdx.x * 8 + wave;
  const float* w = W_fx + (size_t)row * DD;
  float acc = 0.f;
  for (int l = lane; l < DD; l += 32) acc += w[l] * x[l];
  acc = wave_reduce_add(acc);
  if (lane == 0) fxb[row] = acc + b_fx[row] + b_fh[row];
}

// ------- iou = Wx@x + bx + Wh@hsum + bh ; apply gates; o -> out[0:H] -------
__global__ __launch_bounds__(256) void gemv_iou_kernel(
    const float* __restrict__ Wx, const float* __restrict__ bx,
    const float* __restrict__ Wh, const float* __restrict__ bh,
    const float* __restrict__ x,  const float* __restrict__ hsum,
    float* __restrict__ i_g, float* __restrict__ u_g, float* __restrict__ out) {
  int wave = threadIdx.x >> 5, lane = threadIdx.x & 31;
  int row = blockIdx.x * 8 + wave;                 // 0..3H-1
  const float* wxr = Wx + (size_t)row * DD;
  const float* whr = Wh + (size_t)row * HH;
  float acc = 0.f;
  for (int l = lane; l < DD; l += 32) acc += wxr[l] * x[l] + whr[l] * hsum[l];
  acc = wave_reduce_add(acc);
  if (lane == 0) {
    acc += bx[row] + bh[row];
    int gate = row >> 11, idx = row & (HH - 1);
    if (gate == 0)      i_g[idx] = sigmoidf_(acc);
    else if (gate == 1) out[idx] = sigmoidf_(acc);      // o gate
    else                u_g[idx] = tanhf(acc);          // u
  }
}

// ---- fused: fh = ch_h @ W_fh^T ; f = sigmoid(fxb+fh) ; partial_c = sum_m f*ch_c ----
// 256 thr = 8 waves (4x2), tile 128(M) x 128(N), k-step 32, double-buffered LDS.
__global__ __launch_bounds__(256) void fused_gemm_kernel(
    const float* __restrict__ ch_h, const float* __restrict__ ch_c,
    const float* __restrict__ W_fh, const float* __restrict__ fxb,
    float* __restrict__ partials) {
  __shared__ __align__(16) unsigned short lA[2][128][40];  // 80B rows, 16B-aligned slices
  __shared__ __align__(16) unsigned short lB[2][128][40];
  __shared__ float sC[4][128];

  const int tid  = threadIdx.x;
  const int wave = tid >> 5, lane = tid & 31;
  const int wm = wave >> 1, wn = wave & 1;       // wave covers 32(M) x 64(N)
  const int half = lane >> 4, l16 = lane & 15;
  const int n_base = blockIdx.x * 128;
  const int m_base = blockIdx.y * 128;

  // per-thread staging coords: 4 float4 chunks per operand per k-step
  const int srow = tid >> 3;          // 0..31  (+ it*32 rows)
  const int scg  = tid & 7;           // float4 group within 32-wide k slice

  v8f acc[2][4];
  const v8f vzero = {0.f,0.f,0.f,0.f,0.f,0.f,0.f,0.f};
#pragma unroll
  for (int mi = 0; mi < 2; ++mi)
#pragma unroll
    for (int ni = 0; ni < 4; ++ni) acc[mi][ni] = vzero;

  // prologue: global-load tile 0 into registers
  float4 ra[4], rb[4];
#pragma unroll
  for (int it = 0; it < 4; ++it) {
    int row = srow + it * 32;
    ra[it] = *reinterpret_cast<const float4*>(
        ch_h + (size_t)(m_base + row) * HH + scg * 4);
    rb[it] = *reinterpret_cast<const float4*>(
        W_fh + (size_t)(n_base + row) * HH + scg * 4);
  }

  for (int t = 0; t < HH / 32; ++t) {
    const int buf = t & 1;
    const int l0n = (t + 1) * 32;
    __syncthreads();   // readers of this buffer (iter t-2) are done
    // store tile t (convert fp32 -> bf16)
#pragma unroll
    for (int it = 0; it < 4; ++it) {
      int row = srow + it * 32;
      *reinterpret_cast<uint2*>(&lA[buf][row][scg * 4]) = pk4bf(ra[it]);
      *reinterpret_cast<uint2*>(&lB[buf][row][scg * 4]) = pk4bf(rb[it]);
    }
    // issue global loads for tile t+1 — in flight during the WMMA section
    if (t + 1 < HH / 32) {
#pragma unroll
      for (int it = 0; it < 4; ++it) {
        int row = srow + it * 32;
        ra[it] = *reinterpret_cast<const float4*>(
            ch_h + (size_t)(m_base + row) * HH + l0n + scg * 4);
        rb[it] = *reinterpret_cast<const float4*>(
            W_fh + (size_t)(n_base + row) * HH + l0n + scg * 4);
      }
    }
    __syncthreads();

    // fragments: lane half -> K base 0/8; elems 0-7: K=base.. ; 8-15: K=16+base..
    v16bf fa[2], fb[4];
#pragma unroll
    for (int mi = 0; mi < 2; ++mi) {
      int r = wm * 32 + mi * 16 + l16;
      v8bf lo = *reinterpret_cast<const v8bf*>(&lA[buf][r][8 * half]);
      v8bf hi = *reinterpret_cast<const v8bf*>(&lA[buf][r][16 + 8 * half]);
      fa[mi] = __builtin_shufflevector(lo, hi, 0,1,2,3,4,5,6,7,8,9,10,11,12,13,14,15);
    }
#pragma unroll
    for (int ni = 0; ni < 4; ++ni) {
      int r = wn * 64 + ni * 16 + l16;
      v8bf lo = *reinterpret_cast<const v8bf*>(&lB[buf][r][8 * half]);
      v8bf hi = *reinterpret_cast<const v8bf*>(&lB[buf][r][16 + 8 * half]);
      fb[ni] = __builtin_shufflevector(lo, hi, 0,1,2,3,4,5,6,7,8,9,10,11,12,13,14,15);
    }
#pragma unroll
    for (int mi = 0; mi < 2; ++mi)
#pragma unroll
      for (int ni = 0; ni < 4; ++ni)
        acc[mi][ni] = __builtin_amdgcn_wmma_f32_16x16x32_bf16(
            false, fa[mi], false, fb[ni], (short)0, acc[mi][ni], false, false);
  }

  // epilogue: f = sigmoid(fxb[j] + fh[k][j]); csum[j] += f * ch_c[k][j]
  // C/D layout: elem r of v8f in lane L -> row r + 8*(L/16), col L%16
  float csum[4] = {0.f, 0.f, 0.f, 0.f};
#pragma unroll
  for (int ni = 0; ni < 4; ++ni) {
    int j = n_base + wn * 64 + ni * 16 + l16;
    float fxbj = fxb[j];
#pragma unroll
    for (int mi = 0; mi < 2; ++mi) {
#pragma unroll
      for (int r = 0; r < 8; ++r) {
        int k = m_base + wm * 32 + mi * 16 + 8 * half + r;
        float f = sigmoidf_(fxbj + acc[mi][ni][r]);
        csum[ni] += f * ch_c[(size_t)k * HH + j];
      }
    }
  }
#pragma unroll
  for (int ni = 0; ni < 4; ++ni) {
    float tot = csum[ni] + __shfl_xor(csum[ni], 16, 32);  // combine row halves
    if (half == 0) sC[wm][wn * 64 + ni * 16 + l16] = tot;
  }
  __syncthreads();
  if (tid < 128) {
    float s = sC[0][tid] + sC[1][tid] + sC[2][tid] + sC[3][tid];  // fixed order
    partials[(size_t)blockIdx.y * HH + n_base + tid] = s;
  }
}

// ---------------- c = i*u + sum partials ; h = o*tanh(c) ----------------
__global__ __launch_bounds__(256) void finalize_kernel(
    const float* __restrict__ i_g, const float* __restrict__ u_g,
    const float* __restrict__ partials, float* __restrict__ out) {
  int j = blockIdx.x * 256 + threadIdx.x;
  float c = i_g[j] * u_g[j];
#pragma unroll
  for (int mb = 0; mb < 32; ++mb) c += partials[(size_t)mb * HH + j];
  out[HH + j]     = c;
  out[2 * HH + j] = out[j] * tanhf(c);   // out[j] holds o
}

extern "C" void kernel_launch(void* const* d_in, const int* in_sizes, int n_in,
                              void* d_out, int out_size, void* d_ws, size_t ws_size,
                              hipStream_t stream) {
  const float* input  = (const float*)d_in[0];
  const float* ch_c   = (const float*)d_in[1];
  const float* ch_h   = (const float*)d_in[2];
  const float* W_ioux = (const float*)d_in[3];
  const float* b_ioux = (const float*)d_in[4];
  const float* W_iouh = (const float*)d_in[5];
  const float* b_iouh = (const float*)d_in[6];
  const float* W_fx   = (const float*)d_in[7];
  const float* b_fx   = (const float*)d_in[8];
  const float* W_fh   = (const float*)d_in[9];
  const float* b_fh   = (const float*)d_in[10];
  float* out = (float*)d_out;

  float* ws       = (float*)d_ws;
  float* fxb      = ws;                  // [H]
  float* i_g      = ws + HH;             // [H]
  float* u_g      = ws + 2 * HH;         // [H]
  float* hsum     = ws + 3 * HH;         // [H]
  float* hpart    = ws + 4 * HH;         // [8][H]
  float* partials = ws + 12 * HH;        // [32][H]

  hsum_part_kernel<<<dim3(HH / 256, 8), 256, 0, stream>>>(ch_h, hpart);
  hsum_reduce_kernel<<<HH / 256, 256, 0, stream>>>(hpart, hsum);
  gemv_fx_kernel<<<HH / 8, 256, 0, stream>>>(W_fx, b_fx, b_fh, input, fxb);
  gemv_iou_kernel<<<(3 * HH) / 8, 256, 0, stream>>>(
      W_ioux, b_ioux, W_iouh, b_iouh, input, hsum, i_g, u_g, out);
  fused_gemm_kernel<<<dim3(HH / 128, KK / 128), 256, 0, stream>>>(
      ch_h, ch_c, W_fh, fxb, partials);
  finalize_kernel<<<HH / 256, 256, 0, stream>>>(i_g, u_g, partials, out);
}
